// DeeperGCN_59493886984412
// MI455X (gfx1250) — compile-verified, matrix operand order checked by reading, not run on hardware
//
#include <hip/hip_runtime.h>

#define Hc  128
#define H2c 256
#define Lc  4
#define Nc  40000
#define Ec  640000

typedef _Float16 v4h  __attribute__((ext_vector_type(4)));
typedef _Float16 v8h  __attribute__((ext_vector_type(8)));
typedef _Float16 v16h __attribute__((ext_vector_type(16)));
typedef float    v8f  __attribute__((ext_vector_type(8)));

// ---------- helpers ----------
__device__ __forceinline__ float waveSum(float v) {
#pragma unroll
  for (int o = 16; o > 0; o >>= 1) v += __shfl_xor(v, o, 32);
  return v;
}

// monotonic float max via int/uint atomics (single native no-return atomic per side;
// correct with -inf init and mixed signs)
__device__ __forceinline__ void atomicMaxF(float* addr, float val) {
  if (val >= 0.0f) atomicMax((int*)addr, __float_as_int(val));
  else             atomicMin((unsigned int*)addr, __float_as_uint(val));
}

// ---------- weight prep: f32 -> f16, transposed to [Ncols, K] row-major ----------
__global__ void __launch_bounds__(256) prep_w1(const float* __restrict__ W1, _Float16* __restrict__ w1t) {
  int i = blockIdx.x * blockDim.x + threadIdx.x;           // over Lc*H2c*Hc
  if (i >= Lc * H2c * Hc) return;
  int l = i / (H2c * Hc), r = i % (H2c * Hc);
  int n = r / Hc, k = r % Hc;                              // w1t[l][n][k] = W1[l][k][n]
  w1t[i] = (_Float16)W1[(size_t)l * Hc * H2c + (size_t)k * H2c + n];
}
__global__ void __launch_bounds__(256) prep_w2(const float* __restrict__ W2, _Float16* __restrict__ w2t) {
  int i = blockIdx.x * blockDim.x + threadIdx.x;           // over Lc*Hc*H2c
  if (i >= Lc * Hc * H2c) return;
  int l = i / (Hc * H2c), r = i % (Hc * H2c);
  int n = r / H2c, k = r % H2c;                            // w2t[l][n][k] = W2[l][k][n]
  w2t[i] = (_Float16)W2[(size_t)l * H2c * Hc + (size_t)k * Hc + n];
}

// ---------- per-node LayerNorm(H=128)+ReLU, one wave32 per node ----------
__global__ void __launch_bounds__(256) ln_relu_h(const float* __restrict__ x,
                                                 const float* __restrict__ g,
                                                 const float* __restrict__ b,
                                                 float* __restrict__ out) {
  int node = (blockIdx.x * blockDim.x + threadIdx.x) >> 5;
  int lane = threadIdx.x & 31;
  if (node >= Nc) return;
  float4 v = ((const float4*)(x + (size_t)node * Hc))[lane];
  float  s = waveSum(v.x + v.y + v.z + v.w);
  float  mean = s * (1.0f / Hc);
  float4 d = make_float4(v.x - mean, v.y - mean, v.z - mean, v.w - mean);
  float  sq = waveSum(d.x * d.x + d.y * d.y + d.z * d.z + d.w * d.w);
  float  rstd = rsqrtf(sq * (1.0f / Hc) + 1e-5f);
  float4 gv = ((const float4*)g)[lane];
  float4 bv = ((const float4*)b)[lane];
  float4 o;
  o.x = fmaxf(d.x * rstd * gv.x + bv.x, 0.0f);
  o.y = fmaxf(d.y * rstd * gv.y + bv.y, 0.0f);
  o.z = fmaxf(d.z * rstd * gv.z + bv.z, 0.0f);
  o.w = fmaxf(d.w * rstd * gv.w + bv.w, 0.0f);
  ((float4*)(out + (size_t)node * Hc))[lane] = o;
}

// ---------- bias + LayerNorm(2H=256) + ReLU + cast f16, one wave32 per node ----------
__global__ void __launch_bounds__(256) ln_relu_2h_cast(const float* __restrict__ y,
                                                       const float* __restrict__ bias,
                                                       const float* __restrict__ g,
                                                       const float* __restrict__ be,
                                                       _Float16* __restrict__ h16) {
  int node = (blockIdx.x * blockDim.x + threadIdx.x) >> 5;
  int lane = threadIdx.x & 31;
  if (node >= Nc) return;
  const float* row = y + (size_t)node * H2c;
  float4 v0 = ((const float4*)row)[lane * 2 + 0];
  float4 v1 = ((const float4*)row)[lane * 2 + 1];
  float4 bi0 = ((const float4*)bias)[lane * 2 + 0];
  float4 bi1 = ((const float4*)bias)[lane * 2 + 1];
  v0.x += bi0.x; v0.y += bi0.y; v0.z += bi0.z; v0.w += bi0.w;
  v1.x += bi1.x; v1.y += bi1.y; v1.z += bi1.z; v1.w += bi1.w;
  float s = waveSum(v0.x + v0.y + v0.z + v0.w + v1.x + v1.y + v1.z + v1.w);
  float mean = s * (1.0f / H2c);
  float d[8] = {v0.x - mean, v0.y - mean, v0.z - mean, v0.w - mean,
                v1.x - mean, v1.y - mean, v1.z - mean, v1.w - mean};
  float sq = 0.0f;
#pragma unroll
  for (int j = 0; j < 8; ++j) sq += d[j] * d[j];
  sq = waveSum(sq);
  float rstd = rsqrtf(sq * (1.0f / H2c) + 1e-5f);
  float4 g0 = ((const float4*)g)[lane * 2 + 0];
  float4 g1 = ((const float4*)g)[lane * 2 + 1];
  float4 e0 = ((const float4*)be)[lane * 2 + 0];
  float4 e1 = ((const float4*)be)[lane * 2 + 1];
  float gg[8] = {g0.x, g0.y, g0.z, g0.w, g1.x, g1.y, g1.z, g1.w};
  float ee[8] = {e0.x, e0.y, e0.z, e0.w, e1.x, e1.y, e1.z, e1.w};
  v8h o;
#pragma unroll
  for (int j = 0; j < 8; ++j)
    o[j] = (_Float16)fmaxf(d[j] * rstd * gg[j] + ee[j], 0.0f);
  ((v8h*)(h16 + (size_t)node * H2c))[lane] = o;
}

// ---------- init segment buffers (float4) ----------
__global__ void __launch_bounds__(256) init_seg(float* __restrict__ smax,
                                                float* __restrict__ denom,
                                                float* __restrict__ aggn) {
  int i = blockIdx.x * blockDim.x + threadIdx.x;
  if (i >= Nc * Hc / 4) return;
  const float ninf = -__builtin_huge_valf();
  ((float4*)smax)[i]  = make_float4(ninf, ninf, ninf, ninf);
  ((float4*)denom)[i] = make_float4(0.f, 0.f, 0.f, 0.f);
  ((float4*)aggn)[i]  = make_float4(0.f, 0.f, 0.f, 0.f);
}

// ---------- edge passes: 32 lanes per edge, float4 of channels per lane ----------
__device__ __forceinline__ float4 edge_msg(const float* __restrict__ xin, int s, int lane) {
  float4 v = ((const float4*)(xin + (size_t)s * Hc))[lane];
  v.x = fmaxf(v.x, 0.0f) + 1e-7f;
  v.y = fmaxf(v.y, 0.0f) + 1e-7f;
  v.z = fmaxf(v.z, 0.0f) + 1e-7f;
  v.w = fmaxf(v.w, 0.0f) + 1e-7f;
  return v;
}

__global__ void __launch_bounds__(256) edge_max(const float* __restrict__ xin,
                                                const int* __restrict__ src,
                                                const int* __restrict__ dst,
                                                float* __restrict__ smax,
                                                const float* __restrict__ tarr, int layer) {
  int gid = blockIdx.x * blockDim.x + threadIdx.x;
  int e = gid >> 5, lane = gid & 31;
  if (e >= Ec) return;
  int s = src[e], d = dst[e];
  float t = tarr[layer];
  float4 m = edge_msg(xin, s, lane);
  float* sm = smax + (size_t)d * Hc + lane * 4;
  atomicMaxF(sm + 0, m.x * t);
  atomicMaxF(sm + 1, m.y * t);
  atomicMaxF(sm + 2, m.z * t);
  atomicMaxF(sm + 3, m.w * t);
}

// fused pass: denom += ex, aggnum += m*ex  (division by denom deferred to node level,
// valid because denom is segment-constant: sum(m*ex)/denom == sum(m*ex/denom))
__global__ void __launch_bounds__(256) edge_sum(const float* __restrict__ xin,
                                                const int* __restrict__ src,
                                                const int* __restrict__ dst,
                                                const float* __restrict__ smax,
                                                float* __restrict__ denom,
                                                float* __restrict__ aggn,
                                                const float* __restrict__ tarr, int layer) {
  int gid = blockIdx.x * blockDim.x + threadIdx.x;
  int e = gid >> 5, lane = gid & 31;
  if (e >= Ec) return;
  int s = src[e], d = dst[e];
  float t = tarr[layer];
  float4 m  = edge_msg(xin, s, lane);
  float4 sm = ((const float4*)(smax + (size_t)d * Hc))[lane];
  float ex0 = __expf(m.x * t - sm.x);
  float ex1 = __expf(m.y * t - sm.y);
  float ex2 = __expf(m.z * t - sm.z);
  float ex3 = __expf(m.w * t - sm.w);
  float* dn = denom + (size_t)d * Hc + lane * 4;
  float* ag = aggn  + (size_t)d * Hc + lane * 4;
  atomicAdd(dn + 0, ex0);
  atomicAdd(dn + 1, ex1);
  atomicAdd(dn + 2, ex2);
  atomicAdd(dn + 3, ex3);
  atomicAdd(ag + 0, m.x * ex0);
  atomicAdd(ag + 1, m.y * ex1);
  atomicAdd(ag + 2, m.z * ex2);
  atomicAdd(ag + 3, m.w * ex3);
}

// ---------- (aggnum/denom + x) -> f16, float4 vectorized ----------
__global__ void __launch_bounds__(256) agg_cast(const float* __restrict__ aggn,
                                                const float* __restrict__ denom,
                                                const float* __restrict__ xin,
                                                _Float16* __restrict__ a16) {
  int i = blockIdx.x * blockDim.x + threadIdx.x;
  if (i >= Nc * Hc / 4) return;
  float4 an = ((const float4*)aggn)[i];
  float4 dn = ((const float4*)denom)[i];
  float4 xv = ((const float4*)xin)[i];
  v4h o;
  o[0] = (_Float16)(an.x / (dn.x + 1e-16f) + xv.x);
  o[1] = (_Float16)(an.y / (dn.y + 1e-16f) + xv.y);
  o[2] = (_Float16)(an.z / (dn.z + 1e-16f) + xv.z);
  o[3] = (_Float16)(an.w / (dn.w + 1e-16f) + xv.w);
  ((v4h*)a16)[i] = o;
}

// ---------- WMMA GEMM: C[M,NC] = A[M,K] * BT[NC,K]^T, one wave per 16x16 tile ----------
__global__ void __launch_bounds__(256) wmma_gemm(const _Float16* __restrict__ A,
                                                 const _Float16* __restrict__ BT,
                                                 float* __restrict__ C,
                                                 int M, int K, int NC) {
  int wave = (blockIdx.x * (blockDim.x >> 5)) + (threadIdx.x >> 5);
  int lane = threadIdx.x & 31;
  int tilesN = NC >> 4;
  int tilesM = M >> 4;
  int mt = wave / tilesN, nt = wave % tilesN;
  if (mt >= tilesM) return;
  int lr = lane & 15, hi = lane >> 4;
  // ISA 16-bit A/B layout: lanes 0-15 hold K {k0..k0+7, k0+16..k0+23},
  // lanes 16-31 hold the +8-shifted halves; row index = lane%16.
  const _Float16* arow = A  + (size_t)(mt * 16 + lr) * K + hi * 8;
  const _Float16* brow = BT + (size_t)(nt * 16 + lr) * K + hi * 8;
  v8f c = {};
  for (int k0 = 0; k0 < K; k0 += 32) {
    v8h a0 = *(const v8h*)(arow + k0);
    v8h a1 = *(const v8h*)(arow + k0 + 16);
    v8h b0 = *(const v8h*)(brow + k0);
    v8h b1 = *(const v8h*)(brow + k0 + 16);
    v16h a = __builtin_shufflevector(a0, a1, 0,1,2,3,4,5,6,7,8,9,10,11,12,13,14,15);
    v16h b = __builtin_shufflevector(b0, b1, 0,1,2,3,4,5,6,7,8,9,10,11,12,13,14,15);
    c = __builtin_amdgcn_wmma_f32_16x16x32_f16(false, a, false, b, (short)0, c,
                                               false, false);
  }
  // C/D layout: VGPR r -> row mt*16 + r + hi*8, col nt*16 + lane%16
  float* crow = C + (size_t)(mt * 16 + hi * 8) * NC + nt * 16 + lr;
#pragma unroll
  for (int r = 0; r < 8; ++r) crow[(size_t)r * NC] = c[r];
}

// ---------- conv epilogue: x_new = (res ? x_old : 0) + y + b2, float4 ----------
__global__ void __launch_bounds__(256) finish(float* __restrict__ xcur,
                                              const float* __restrict__ y,
                                              const float* __restrict__ b2, int res) {
  int i = blockIdx.x * blockDim.x + threadIdx.x;
  if (i >= Nc * Hc / 4) return;
  float4 yv = ((const float4*)y)[i];
  float4 bv = ((const float4*)b2)[i & (Hc / 4 - 1)];
  float4 base = make_float4(0.f, 0.f, 0.f, 0.f);
  if (res) base = ((const float4*)xcur)[i];
  float4 o = make_float4(base.x + yv.x + bv.x, base.y + yv.y + bv.y,
                         base.z + yv.z + bv.z, base.w + yv.w + bv.w);
  ((float4*)xcur)[i] = o;
}

// ---------- host ----------
extern "C" void kernel_launch(void* const* d_in, const int* in_sizes, int n_in,
                              void* d_out, int out_size, void* d_ws, size_t ws_size,
                              hipStream_t stream) {
  (void)in_sizes; (void)n_in; (void)out_size; (void)ws_size;
  const float* x   = (const float*)d_in[0];
  const int*   ei  = (const int*)d_in[1];
  const int*   src = ei;
  const int*   dst = ei + Ec;
  const float* W1  = (const float*)d_in[2];
  const float* b1  = (const float*)d_in[3];
  const float* g1  = (const float*)d_in[4];
  const float* be1 = (const float*)d_in[5];
  const float* W2  = (const float*)d_in[6];
  const float* b2  = (const float*)d_in[7];
  const float* t   = (const float*)d_in[8];
  const float* lng = (const float*)d_in[9];
  const float* lnb = (const float*)d_in[10];
  float* out = (float*)d_out;

  const size_t NH  = (size_t)Nc * Hc * sizeof(float);      // 20,480,000 B
  char* ws = (char*)d_ws;
  float* x_cur = (float*)ws;
  float* hpre  = (float*)(ws + NH);
  char*  B     = ws + 2 * NH;                              // 61.44 MB aliased region
  float* smax  = (float*)B;                                // edge phase
  float* denom = (float*)(B + NH);
  float* aggn  = (float*)(B + 2 * NH);
  float* out1  = (float*)B;                                // gemm phase (aliases smax/denom)
  _Float16* h16 = (_Float16*)(B + 2 * NH);                 // aliases aggn
  float* out2  = (float*)B;                                // aliases out1 (dead after ln2)
  _Float16* a16 = (_Float16*)(ws + 5 * NH);
  _Float16* w1t = (_Float16*)(ws + 5 * NH + (size_t)Nc * Hc * sizeof(_Float16));
  _Float16* w2t = w1t + (size_t)Lc * H2c * Hc;

  dim3 blk(256);
  int wBlk   = (Lc * H2c * Hc + 255) / 256;
  int node4B = (Nc * Hc / 4 + 255) / 256;                  // float4 elementwise
  int waveB  = (Nc * 32 + 255) / 256;                      // one wave per node
  int edgeB  = (Ec * 32 + 255) / 256;                      // 32 lanes per edge
  int gemm1B = ((Nc / 16) * (H2c / 16) + 7) / 8;           // 8 waves per block
  int gemm2B = ((Nc / 16) * (Hc  / 16) + 7) / 8;

  prep_w1<<<wBlk, blk, 0, stream>>>(W1, w1t);
  prep_w2<<<wBlk, blk, 0, stream>>>(W2, w2t);

  for (int l = 0; l < Lc; ++l) {
    const float* cin;
    if (l == 0) {
      cin = x;
    } else {
      ln_relu_h<<<waveB, blk, 0, stream>>>(x_cur, lng + l * Hc, lnb + l * Hc, hpre);
      cin = hpre;
    }
    init_seg<<<node4B, blk, 0, stream>>>(smax, denom, aggn);
    edge_max<<<edgeB, blk, 0, stream>>>(cin, src, dst, smax, t, l);
    edge_sum<<<edgeB, blk, 0, stream>>>(cin, src, dst, smax, denom, aggn, t, l);
    agg_cast<<<node4B, blk, 0, stream>>>(aggn, denom, cin, a16);
    wmma_gemm<<<gemm1B, blk, 0, stream>>>(a16, w1t + (size_t)l * H2c * Hc, out1,
                                          Nc, Hc, H2c);
    ln_relu_2h_cast<<<waveB, blk, 0, stream>>>(out1, b1 + l * H2c, g1 + l * H2c,
                                               be1 + l * H2c, h16);
    wmma_gemm<<<gemm2B, blk, 0, stream>>>(h16, w2t + (size_t)l * Hc * H2c, out2,
                                          Nc, H2c, Hc);
    finish<<<node4B, blk, 0, stream>>>(x_cur, out2, b2 + l * Hc, l > 0);
  }
  // epilogue: relu(LN(x, ln_g[0], ln_b[0]))
  ln_relu_h<<<waveB, blk, 0, stream>>>(x_cur, lng, lnb, out);
}
